// AdaDyGNN_78795470012526
// MI455X (gfx1250) — matrix operand order repaired and checked by reading, not compiled, not run on hardware
//
#include <hip/hip_runtime.h>
#include <hip/hip_bf16.h>
#include <math.h>

// ---------------- problem constants (from reference) ----------------
constexpr int NN = 500000;   // N_NODES
constexpr int BB = 4096;     // batch
constexpr int KK = 20;       // n_neighbors
constexpr int UU = 10;       // n_update_neighbors
constexpr int EE = 128;      // emb dim == MD

typedef __attribute__((ext_vector_type(16))) _Float16 v16h;
typedef __attribute__((ext_vector_type(8)))  _Float16 v8h;
typedef __attribute__((ext_vector_type(4)))  _Float16 v4h;
typedef __attribute__((ext_vector_type(8)))  float    v8f;

#define WMMA_F16(Afrag, Bfrag, Cacc) \
  __builtin_amdgcn_wmma_f32_16x16x32_f16(false, (Afrag), false, (Bfrag), (short)0, (Cacc), false, false)

// ---- WMMA fragment loaders (ISA 05_wmma layouts, wave32) ----
// A: 16x32 f16. Lane holds row M = lane%16, K-halves:
//   base=(lane<16?0:8); halves 0..7 -> K=base+0..7 ; halves 8..15 -> K=base+16..23
__device__ __forceinline__ v16h load_a_frag(const _Float16* As, int ld, int mBase, int kBase, int lane) {
  const _Float16* row = As + (size_t)(mBase + (lane & 15)) * ld + kBase + ((lane & 16) ? 8 : 0);
  v8h lo = *(const v8h*)(row);
  v8h hi = *(const v8h*)(row + 16);
  return __builtin_shufflevector(lo, hi, 0, 1, 2, 3, 4, 5, 6, 7, 8, 9, 10, 11, 12, 13, 14, 15);
}
// B: 32x16 f16 staged TRANSPOSED in LDS as Bt[n][k]: lane's K split like A (base+0..7, base+16..23).
__device__ __forceinline__ v16h load_bt_frag(const _Float16* Bt, int ldk, int kBase, int nBase, int lane) {
  const _Float16* p = Bt + (size_t)(nBase + (lane & 15)) * ldk + kBase + ((lane & 16) ? 8 : 0);
  v8h lo = *(const v8h*)(p);
  v8h hi = *(const v8h*)(p + 16);
  return __builtin_shufflevector(lo, hi, 0, 1, 2, 3, 4, 5, 6, 7, 8, 9, 10, 11, 12, 13, 14, 15);
}
// C/D f32 16x16: VGPR v -> row = v + (lane>=16 ? 8 : 0), col = lane%16

__device__ __forceinline__ float sigmoidf_(float x) { return 1.0f / (1.0f + __expf(-x)); }
__device__ __forceinline__ v4h pack4(float a, float b, float c, float d) {
  v4h h = {(_Float16)a, (_Float16)b, (_Float16)c, (_Float16)d};
  return h;
}

// ==================================================================
// Stage 1: messages (src & dst).  One block per (side,b).
// ==================================================================
__global__ void msg_kernel(const float* __restrict__ mem,
                           const int* __restrict__ nbrs_s, const float* __restrict__ tim_s,
                           const int* __restrict__ nbrs_d, const float* __restrict__ tim_d,
                           const int* __restrict__ sidx, const int* __restrict__ didx,
                           const float* __restrict__ ts, const float* __restrict__ Wg,
                           const float* __restrict__ avec, float* __restrict__ hsum) {
  __shared__ __align__(32) _Float16 As[32][136];   // 128 + 8 pad
  __shared__ __align__(32) _Float16 Bt[64][144];   // transposed W_g, 128 + 16 pad
  __shared__ float    Cs[32][64];
  __shared__ float    lg[21];

  int side = blockIdx.x / BB;
  int b    = blockIdx.x % BB;
  const int*   nbrs = side ? nbrs_d : nbrs_s;
  const float* tim  = side ? tim_d  : tim_s;
  const int*   idxs = side ? didx   : sidx;
  float t = ts[b];
  int tid = threadIdx.x;

  // A rows 0..20: vectorized 16B gathers (rows are 512B aligned)
  for (int i = tid; i < 21 * 32; i += 64) {
    int r = i >> 5, c4 = i & 31;
    float4 v; float d;
    if (r < KK) {
      int node = nbrs[b * KK + r];
      const float4* src = (const float4*)(mem + (size_t)node * EE);
      if (c4 == 0) __builtin_prefetch(src, 0, 1);
      v = src[c4];
      d = 1.0f / (1.0f + (t - tim[b * KK + r]));
    } else {
      v = ((const float4*)(mem + (size_t)idxs[b] * EE))[c4];
      d = 1.0f;
    }
    *(v4h*)&As[r][c4 * 4] = pack4(v.x * d, v.y * d, v.z * d, v.w * d);
  }
  // zero rows 21..31 (16B stores)
  {
    v8h z = {0, 0, 0, 0, 0, 0, 0, 0};
    for (int i = tid; i < 11 * 16; i += 64) {
      int r = 21 + (i >> 4), c8 = i & 15;
      *(v8h*)&As[r][c8 * 8] = z;
    }
  }
  // W_g[k][n] (k<128, n<64) -> transposed Bt[n][k]; one v4h LDS store per 4 k's
  {
    int n = tid & 63;
    for (int j = 0; j < 32; ++j) {
      *(v4h*)&Bt[n][4 * j] = pack4(Wg[(4 * j + 0) * 64 + n], Wg[(4 * j + 1) * 64 + n],
                                   Wg[(4 * j + 2) * 64 + n], Wg[(4 * j + 3) * 64 + n]);
    }
  }
  __syncthreads();

  int lane = tid & 31, wave = tid >> 5;
  int mb = wave * 16;
  v16h afr[4];
#pragma unroll
  for (int kk = 0; kk < 4; ++kk) afr[kk] = load_a_frag(&As[0][0], 136, mb, kk * 32, lane);
  for (int nt = 0; nt < 4; ++nt) {
    v8f acc = {0, 0, 0, 0, 0, 0, 0, 0};
#pragma unroll
    for (int kk = 0; kk < 4; ++kk) {
      v16h bf = load_bt_frag(&Bt[0][0], 144, kk * 32, nt * 16, lane);
      acc = WMMA_F16(afr[kk], bf, acc);
    }
    int rb = mb + ((lane & 16) ? 8 : 0);
    int col = nt * 16 + (lane & 15);
#pragma unroll
    for (int v = 0; v < 8; ++v) Cs[rb + v][col] = acc[v];
  }
  __syncthreads();

  if (tid < KK) {
    float s = 0.0f;
    for (int n = 0; n < 64; ++n) s += Cs[tid][n] * avec[64 + n];
    lg[tid] = s;
  } else if (tid == KK) {
    float s = 0.0f;
    for (int n = 0; n < 64; ++n) s += Cs[20][n] * avec[n];
    lg[20] = s;
  }
  __syncthreads();

  float cb = lg[20];
  float l[KK];
  float mx = -1e30f;
#pragma unroll
  for (int k = 0; k < KK; ++k) {
    float x = cb + lg[k];
    x = x > 0.0f ? x : 0.2f * x;       // leaky_relu(0.2)
    l[k] = x; mx = fmaxf(mx, x);
  }
  float den = 0.0f;
#pragma unroll
  for (int k = 0; k < KK; ++k) { l[k] = __expf(l[k] - mx); den += l[k]; }
  float out = 0.0f;
#pragma unroll
  for (int k = 0; k < KK; ++k) out += (l[k] / den) * Cs[k][tid];
  hsum[(size_t)b * 128 + side * 64 + tid] = out;   // pre-tanh message sum
}

// ==================================================================
// Stage 2: h_cat[:,0:128] = tanh(hsum); h_cat[:,128:256] = tanh(edge_feat[eidx] @ W_e)
// ==================================================================
__global__ void hcat_kernel(const float* __restrict__ ef, const int* __restrict__ eidx,
                            const float* __restrict__ We, const float* __restrict__ hsum,
                            float* __restrict__ hcat) {
  __shared__ __align__(32) _Float16 As[16][136];
  __shared__ __align__(32) _Float16 Bt[128][144];  // transposed W_e
  int b0 = blockIdx.x * 16;
  int tid = threadIdx.x;
  for (int i = tid; i < 16 * 32; i += 128) {
    int r = i >> 5, c4 = i & 31;
    float4 v = ((const float4*)(ef + (size_t)eidx[b0 + r] * 128))[c4];
    *(v4h*)&As[r][c4 * 4] = pack4(v.x, v.y, v.z, v.w);
  }
  {
    int n = tid & 127;
    for (int j = 0; j < 32; ++j) {
      *(v4h*)&Bt[n][4 * j] = pack4(We[(4 * j + 0) * 128 + n], We[(4 * j + 1) * 128 + n],
                                   We[(4 * j + 2) * 128 + n], We[(4 * j + 3) * 128 + n]);
    }
  }
  for (int i = tid; i < 16 * 32; i += 128) {
    int r = i >> 5, c4 = i & 31;
    float4 v = ((const float4*)(hsum + (size_t)(b0 + r) * 128))[c4];
    float4 o = {tanhf(v.x), tanhf(v.y), tanhf(v.z), tanhf(v.w)};
    ((float4*)(hcat + (size_t)(b0 + r) * 256))[c4] = o;
  }
  __syncthreads();
  int lane = tid & 31, wave = tid >> 5;
  v16h afr[4];
#pragma unroll
  for (int kk = 0; kk < 4; ++kk) afr[kk] = load_a_frag(&As[0][0], 136, 0, kk * 32, lane);
  for (int q = 0; q < 2; ++q) {
    int nt = wave * 2 + q;
    v8f acc = {0, 0, 0, 0, 0, 0, 0, 0};
#pragma unroll
    for (int kk = 0; kk < 4; ++kk) {
      v16h bf = load_bt_frag(&Bt[0][0], 144, kk * 32, nt * 16, lane);
      acc = WMMA_F16(afr[kk], bf, acc);
    }
    int rb = (lane & 16) ? 8 : 0;
    int col = nt * 16 + (lane & 15);
#pragma unroll
    for (int v = 0; v < 8; ++v)
      hcat[(size_t)(b0 + rb + v) * 256 + 128 + col] = tanhf(acc[v]);
  }
}

// ==================================================================
// Stage 3: new_src/new_dst = tanh([mem[idx], h_cat] @ W_uc)  (K = 384, 3 chunks)
// ==================================================================
__global__ void uc_kernel(const float* __restrict__ mem, const int* __restrict__ sidx,
                          const int* __restrict__ didx, const float* __restrict__ hcat,
                          const float* __restrict__ Wuc, float* __restrict__ nsrc,
                          float* __restrict__ ndst) {
  __shared__ __align__(32) _Float16 As[16][136];
  __shared__ __align__(32) _Float16 Bt[128][144];
  const int nb = BB / 16;
  int side = blockIdx.x / nb;
  int b0 = (blockIdx.x % nb) * 16;
  const int* idxs = side ? didx : sidx;
  int tid = threadIdx.x, lane = tid & 31, wave = tid >> 5;
  v8f acc[2] = {{0,0,0,0,0,0,0,0},{0,0,0,0,0,0,0,0}};
  for (int kc = 0; kc < 3; ++kc) {
    for (int i = tid; i < 16 * 32; i += 128) {
      int r = i >> 5, c4 = i & 31;
      float4 v = (kc == 0)
          ? ((const float4*)(mem + (size_t)idxs[b0 + r] * EE))[c4]
          : ((const float4*)(hcat + (size_t)(b0 + r) * 256 + (kc - 1) * 128))[c4];
      *(v4h*)&As[r][c4 * 4] = pack4(v.x, v.y, v.z, v.w);
    }
    {
      int n = tid & 127;
      const float* W = Wuc + (size_t)kc * 128 * 128;
      for (int j = 0; j < 32; ++j) {
        *(v4h*)&Bt[n][4 * j] = pack4(W[(4 * j + 0) * 128 + n], W[(4 * j + 1) * 128 + n],
                                     W[(4 * j + 2) * 128 + n], W[(4 * j + 3) * 128 + n]);
      }
    }
    __syncthreads();
    v16h afr[4];
#pragma unroll
    for (int kk = 0; kk < 4; ++kk) afr[kk] = load_a_frag(&As[0][0], 136, 0, kk * 32, lane);
    for (int q = 0; q < 2; ++q) {
      int nt = wave * 2 + q;
#pragma unroll
      for (int kk = 0; kk < 4; ++kk) {
        v16h bf = load_bt_frag(&Bt[0][0], 144, kk * 32, nt * 16, lane);
        acc[q] = WMMA_F16(afr[kk], bf, acc[q]);
      }
    }
    __syncthreads();
  }
  float* out = side ? ndst : nsrc;
  for (int q = 0; q < 2; ++q) {
    int nt = wave * 2 + q;
    int rb = (lane & 16) ? 8 : 0;
    int col = nt * 16 + (lane & 15);
#pragma unroll
    for (int v = 0; v < 8; ++v) out[(size_t)(b0 + rb + v) * 128 + col] = tanhf(acc[q][v]);
  }
}

// ==================================================================
// Stage 4: h1 = h_cat @ W_p  (K = 256, 2 chunks, no tanh)
// ==================================================================
__global__ void h1_gemm_kernel(const float* __restrict__ hcat, const float* __restrict__ Wp,
                               float* __restrict__ h1buf) {
  __shared__ __align__(32) _Float16 As[16][136];
  __shared__ __align__(32) _Float16 Bt[128][144];
  int b0 = blockIdx.x * 16;
  int tid = threadIdx.x, lane = tid & 31, wave = tid >> 5;
  v8f acc[2] = {{0,0,0,0,0,0,0,0},{0,0,0,0,0,0,0,0}};
  for (int kc = 0; kc < 2; ++kc) {
    for (int i = tid; i < 16 * 32; i += 128) {
      int r = i >> 5, c4 = i & 31;
      float4 v = ((const float4*)(hcat + (size_t)(b0 + r) * 256 + kc * 128))[c4];
      *(v4h*)&As[r][c4 * 4] = pack4(v.x, v.y, v.z, v.w);
    }
    {
      int n = tid & 127;
      const float* W = Wp + (size_t)kc * 128 * 128;
      for (int j = 0; j < 32; ++j) {
        *(v4h*)&Bt[n][4 * j] = pack4(W[(4 * j + 0) * 128 + n], W[(4 * j + 1) * 128 + n],
                                     W[(4 * j + 2) * 128 + n], W[(4 * j + 3) * 128 + n]);
      }
    }
    __syncthreads();
    v16h afr[4];
#pragma unroll
    for (int kk = 0; kk < 4; ++kk) afr[kk] = load_a_frag(&As[0][0], 136, 0, kk * 32, lane);
    for (int q = 0; q < 2; ++q) {
      int nt = wave * 2 + q;
#pragma unroll
      for (int kk = 0; kk < 4; ++kk) {
        v16h bf = load_bt_frag(&Bt[0][0], 144, kk * 32, nt * 16, lane);
        acc[q] = WMMA_F16(afr[kk], bf, acc[q]);
      }
    }
    __syncthreads();
  }
  for (int q = 0; q < 2; ++q) {
    int nt = wave * 2 + q;
    int rb = (lane & 16) ? 8 : 0;
    int col = nt * 16 + (lane & 15);
#pragma unroll
    for (int v = 0; v < 8; ++v) h1buf[(size_t)(b0 + rb + v) * 128 + col] = acc[q][v];
  }
}

// ---------------- deterministic scatter machinery ----------------
__global__ void tag_reset(int* __restrict__ tag) {
  int i = blockIdx.x * 256 + threadIdx.x;
  if (i < NN) tag[i] = -1;
}
__global__ void tag_max(const int* __restrict__ idxs, int n, int* __restrict__ tag) {
  int i = blockIdx.x * 256 + threadIdx.x;
  if (i < n) atomicMax(&tag[idxs[i]], i);
}
// 16B-vectorized row scatter (winner row only)
__global__ void row_scatter(const float* __restrict__ rows, const int* __restrict__ idxs,
                            const int* __restrict__ tag, int n, float* __restrict__ wmem) {
  int g = blockIdx.x * 256 + threadIdx.x;
  if (g >= n * 32) return;
  int i = g >> 5, c4 = g & 31;
  int node = idxs[i];
  if (tag[node] == i)
    ((float4*)(wmem + (size_t)node * EE))[c4] = ((const float4*)(rows + (size_t)i * 128))[c4];
}

// ==================================================================
// Update stage A: attention over U neighbors; changed = h1 * att
// One block per b, 10 waves (one per u); 16B loads per lane.
// ==================================================================
__global__ void att_kernel(const float* __restrict__ wmem, const int* __restrict__ nbrs,
                           const float* __restrict__ tim, const float* __restrict__ ts,
                           const float* __restrict__ h1buf, float* __restrict__ chg) {
  __shared__ float lg[UU];
  __shared__ float att[UU];
  int b = blockIdx.x;
  int tid = threadIdx.x, lane = tid & 31, u = tid >> 5;
  float t = ts[b];
  {
    int node = nbrs[b * UU + u];
    float d = 1.0f / (1.0f + (t - tim[b * UU + u]));
    float4 hv = ((const float4*)(h1buf + (size_t)b * 128))[lane];
    float4 mv = ((const float4*)(wmem + (size_t)node * EE))[lane];
    float e0 = hv.x * mv.x * d, e1 = hv.y * mv.y * d;
    float e2 = hv.z * mv.z * d, e3 = hv.w * mv.w * d;
    float s  = e0 + e1 + e2 + e3;
    float nn = e0 * e0 + e1 * e1 + e2 * e2 + e3 * e3;
    for (int off = 16; off; off >>= 1) {
      s  += __shfl_down(s,  off, 32);
      nn += __shfl_down(nn, off, 32);
    }
    if (lane == 0) lg[u] = s / (sqrtf(nn) + 1e-10f);
  }
  __syncthreads();
  if (tid == 0) {
    float mx = -1e30f;
    for (int k = 0; k < UU; ++k) mx = fmaxf(mx, lg[k]);
    float den = 0.0f, e[UU];
    for (int k = 0; k < UU; ++k) { e[k] = __expf(lg[k] - mx); den += e[k]; }
    for (int k = 0; k < UU; ++k) att[k] = e[k] / den;
  }
  __syncthreads();
  for (int i = tid; i < UU * 32; i += 320) {
    int uu = i >> 5, c4 = i & 31;
    float4 hv = ((const float4*)(h1buf + (size_t)b * 128))[c4];
    float a = att[uu];
    float4 o = {hv.x * a, hv.y * a, hv.z * a, hv.w * a};
    ((float4*)(chg + ((size_t)b * UU + uu) * 128))[c4] = o;
  }
}

// ==================================================================
// Update stage B: fused dual GEMM over M = B*U rows (K = 256, 4 chunks of 64):
//   y   = relu([nemb, changed] @ W_1) ; p = sigmoid(y @ W_2) ; mask = p>=0.5
//   new = tanh([old,  changed] @ W_un)
// ==================================================================
__global__ void upgemm_kernel(const float* __restrict__ wmem, const int* __restrict__ nbrs,
                              const float* __restrict__ tim, const float* __restrict__ ts,
                              const float* __restrict__ chg, const float* __restrict__ W1,
                              const float* __restrict__ W2, const float* __restrict__ Wun,
                              float* __restrict__ nup, float* __restrict__ maskf) {
  __shared__ __align__(32) _Float16 A1[16][72];    // 64 + 8 pad
  __shared__ __align__(32) _Float16 A2[16][72];
  __shared__ __align__(32) _Float16 B1t[128][80];  // transposed chunk, 64 + 16 pad
  __shared__ __align__(32) _Float16 B2t[128][80];
  __shared__ float    Ys[16][128];
  int g0 = blockIdx.x * 16;
  int tid = threadIdx.x, lane = tid & 31, wave = tid >> 5;
  v8f acc1[2] = {{0,0,0,0,0,0,0,0},{0,0,0,0,0,0,0,0}};
  v8f acc2[2] = {{0,0,0,0,0,0,0,0},{0,0,0,0,0,0,0,0}};
  for (int kc = 0; kc < 4; ++kc) {
    for (int i = tid; i < 16 * 16; i += 128) {
      int r = i >> 4, c4 = i & 15;
      int gr = g0 + r, b = gr / UU, u = gr % UU;
      if (kc < 2) {
        int node = nbrs[b * UU + u];
        float4 v = ((const float4*)(wmem + (size_t)node * EE))[kc * 16 + c4];
        float d = 1.0f / (1.0f + (ts[b] - tim[b * UU + u]));
        *(v4h*)&A1[r][c4 * 4] = pack4(v.x * d, v.y * d, v.z * d, v.w * d); // nemb
        *(v4h*)&A2[r][c4 * 4] = pack4(v.x, v.y, v.z, v.w);                 // old
      } else {
        float4 v = ((const float4*)(chg + (size_t)gr * 128 + (kc - 2) * 64))[c4];
        v4h h = pack4(v.x, v.y, v.z, v.w);
        *(v4h*)&A1[r][c4 * 4] = h;
        *(v4h*)&A2[r][c4 * 4] = h;
      }
    }
    {
      int n = tid & 127;
      const float* Wa = W1  + (size_t)kc * 64 * 128;
      const float* Wb = Wun + (size_t)kc * 64 * 128;
      for (int j = 0; j < 16; ++j) {
        *(v4h*)&B1t[n][4 * j] = pack4(Wa[(4 * j + 0) * 128 + n], Wa[(4 * j + 1) * 128 + n],
                                      Wa[(4 * j + 2) * 128 + n], Wa[(4 * j + 3) * 128 + n]);
        *(v4h*)&B2t[n][4 * j] = pack4(Wb[(4 * j + 0) * 128 + n], Wb[(4 * j + 1) * 128 + n],
                                      Wb[(4 * j + 2) * 128 + n], Wb[(4 * j + 3) * 128 + n]);
      }
    }
    __syncthreads();
    v16h a1f[2], a2f[2];
#pragma unroll
    for (int kk = 0; kk < 2; ++kk) {
      a1f[kk] = load_a_frag(&A1[0][0], 72, 0, kk * 32, lane);
      a2f[kk] = load_a_frag(&A2[0][0], 72, 0, kk * 32, lane);
    }
    for (int q = 0; q < 2; ++q) {
      int nt = wave * 2 + q;
#pragma unroll
      for (int kk = 0; kk < 2; ++kk) {
        v16h b1 = load_bt_frag(&B1t[0][0], 80, kk * 32, nt * 16, lane);
        acc1[q] = WMMA_F16(a1f[kk], b1, acc1[q]);
        v16h b2 = load_bt_frag(&B2t[0][0], 80, kk * 32, nt * 16, lane);
        acc2[q] = WMMA_F16(a2f[kk], b2, acc2[q]);
      }
    }
    __syncthreads();
  }
  for (int q = 0; q < 2; ++q) {
    int nt = wave * 2 + q;
    int rb = (lane & 16) ? 8 : 0;
    int col = nt * 16 + (lane & 15);
#pragma unroll
    for (int v = 0; v < 8; ++v) {
      Ys[rb + v][col] = fmaxf(acc1[q][v], 0.0f);
      nup[(size_t)(g0 + rb + v) * 128 + col] = tanhf(acc2[q][v]);
    }
  }
  __syncthreads();
  if (tid < 16) {
    float s = 0.0f;
    for (int n = 0; n < 128; ++n) s += Ys[tid][n] * W2[n];
    maskf[g0 + tid] = (sigmoidf_(s) >= 0.5f) ? 1.0f : 0.0f;
  }
}

__global__ void up_scatter(const float* __restrict__ nup, const int* __restrict__ nbrs,
                           const int* __restrict__ tag, const float* __restrict__ maskf,
                           float* __restrict__ wmem) {
  int g = blockIdx.x * 256 + threadIdx.x;
  if (g >= BB * UU * 32) return;
  int i = g >> 5, c4 = g & 31;
  int node = nbrs[i];
  if (tag[node] == i && maskf[i] > 0.5f)
    ((float4*)(wmem + (size_t)node * EE))[c4] = ((const float4*)(nup + (size_t)i * 128))[c4];
}

// ==================================================================
// Final: pos/neg scores (one wave per batch row, 16B loads)
// ==================================================================
__global__ void score_kernel(const float* __restrict__ wmem, const int* __restrict__ sidx,
                             const int* __restrict__ didx, const int* __restrict__ nidx,
                             float* __restrict__ out) {
  int b = blockIdx.x * (blockDim.x / 32) + (threadIdx.x >> 5);
  int lane = threadIdx.x & 31;
  if (b >= BB) return;
  float4 s4 = ((const float4*)(wmem + (size_t)sidx[b] * EE))[lane];
  float4 d4 = ((const float4*)(wmem + (size_t)didx[b] * EE))[lane];
  float4 n4 = ((const float4*)(wmem + (size_t)nidx[b] * EE))[lane];
  float a = s4.x * d4.x + s4.y * d4.y + s4.z * d4.z + s4.w * d4.w;
  float c = s4.x * n4.x + s4.y * n4.y + s4.z * n4.z + s4.w * n4.w;
  for (int off = 16; off; off >>= 1) {
    a += __shfl_down(a, off, 32);
    c += __shfl_down(c, off, 32);
  }
  if (lane == 0) {
    out[b]      = sigmoidf_(a);
    out[BB + b] = sigmoidf_(c);
  }
}

// ==================================================================
extern "C" void kernel_launch(void* const* d_in, const int* in_sizes, int n_in,
                              void* d_out, int out_size, void* d_ws, size_t ws_size,
                              hipStream_t stream) {
  (void)in_sizes; (void)n_in; (void)out_size; (void)ws_size;
  const float* mem    = (const float*)d_in[0];
  const float* ef     = (const float*)d_in[1];
  const int*   sidx   = (const int*)d_in[2];
  const int*   didx   = (const int*)d_in[3];
  const int*   nidx   = (const int*)d_in[4];
  const int*   eidx   = (const int*)d_in[5];
  const float* ts     = (const float*)d_in[6];
  const int*   nbrs_s = (const int*)d_in[7];
  const float* tim_s  = (const float*)d_in[8];
  const int*   nbrs_d = (const int*)d_in[9];
  const float* tim_d  = (const float*)d_in[10];
  const int*   unb_s  = (const int*)d_in[11];
  const float* utim_s = (const float*)d_in[12];
  const int*   unb_d  = (const int*)d_in[13];
  const float* utim_d = (const float*)d_in[14];
  const float* Wg     = (const float*)d_in[15];
  const float* avec   = (const float*)d_in[16];
  const float* We     = (const float*)d_in[17];
  const float* Wuc    = (const float*)d_in[18];
  const float* Wun    = (const float*)d_in[19];
  const float* Wp     = (const float*)d_in[20];
  const float* W1     = (const float*)d_in[21];
  const float* W2     = (const float*)d_in[22];

  float* ws = (float*)d_ws;
  size_t o = 0;
  float* wmem  = ws + o; o += (size_t)NN * EE;       // mutable memory copy
  float* hsum  = ws + o; o += (size_t)BB * 128;      // pre-tanh message sums
  float* hcat  = ws + o; o += (size_t)BB * 256;      // [h, h_e]
  float* h1buf = ws + o; o += (size_t)BB * 128;      // h @ W_p
  float* nsrc  = ws + o; o += (size_t)BB * 128;
  float* ndst  = ws + o; o += (size_t)BB * 128;
  float* chg   = ws + o; o += (size_t)BB * UU * 128; // changed rows
  float* nup   = ws + o; o += (size_t)BB * UU * 128; // candidate new rows
  float* maskf = ws + o; o += (size_t)BB * UU;
  int*   tag   = (int*)(ws + o);                     // NN ints

  // 0) mutable copy of memory table (inputs must stay pristine)
  hipMemcpyAsync(wmem, mem, (size_t)NN * EE * sizeof(float),
                 hipMemcpyDeviceToDevice, stream);

  // 1) messages (reads ORIGINAL mem) -> hsum
  msg_kernel<<<2 * BB, 64, 0, stream>>>(mem, nbrs_s, tim_s, nbrs_d, tim_d,
                                        sidx, didx, ts, Wg, avec, hsum);
  // 2) h_cat
  hcat_kernel<<<BB / 16, 128, 0, stream>>>(ef, eidx, We, hsum, hcat);
  // 3) candidate rows for src/dst (reads ORIGINAL mem rows)
  uc_kernel<<<2 * (BB / 16), 128, 0, stream>>>(mem, sidx, didx, hcat, Wuc, nsrc, ndst);
  // 4) h1 = h @ W_p
  h1_gemm_kernel<<<BB / 16, 128, 0, stream>>>(hcat, Wp, h1buf);

  const int tagBlocks = (NN + 255) / 256;
  // 5) scatter src rows, then dst rows (dst overwrites, matching reference)
  tag_reset<<<tagBlocks, 256, 0, stream>>>(tag);
  tag_max<<<(BB + 255) / 256, 256, 0, stream>>>(sidx, BB, tag);
  row_scatter<<<(BB * 32 + 255) / 256, 256, 0, stream>>>(nsrc, sidx, tag, BB, wmem);
  tag_reset<<<tagBlocks, 256, 0, stream>>>(tag);
  tag_max<<<(BB + 255) / 256, 256, 0, stream>>>(didx, BB, tag);
  row_scatter<<<(BB * 32 + 255) / 256, 256, 0, stream>>>(ndst, didx, tag, BB, wmem);

  // 6) update pass 1 (src update-neighbors), then pass 2 (dst)
  const int* unb[2]    = {unb_s, unb_d};
  const float* utim[2] = {utim_s, utim_d};
  for (int pass = 0; pass < 2; ++pass) {
    att_kernel<<<BB, 320, 0, stream>>>(wmem, unb[pass], utim[pass], ts, h1buf, chg);
    upgemm_kernel<<<(BB * UU) / 16, 128, 0, stream>>>(wmem, unb[pass], utim[pass], ts,
                                                      chg, W1, W2, Wun, nup, maskf);
    tag_reset<<<tagBlocks, 256, 0, stream>>>(tag);
    tag_max<<<(BB * UU + 255) / 256, 256, 0, stream>>>(unb[pass], BB * UU, tag);
    up_scatter<<<(BB * UU * 32 + 255) / 256, 256, 0, stream>>>(nup, unb[pass], tag,
                                                               maskf, wmem);
  }

  // 7) scores
  score_kernel<<<BB / 8, 256, 0, stream>>>(wmem, sidx, didx, nidx, (float*)d_out);
}
